// GRU_50989851738666
// MI455X (gfx1250) — matrix operand order repaired and checked by reading, not compiled, not run on hardware
//
#include <hip/hip_runtime.h>
#include <hip/hip_bf16.h>

#define TT 4096
#define EE 2048
#define HH 2048

#define NBLK   64     // persistent blocks for the scan (low LDS -> residency safe)
#define ROWS   32     // HH / NBLK rows of h owned per block
#define THREADS 256

typedef __attribute__((ext_vector_type(16))) __bf16 v16bf;
typedef __attribute__((ext_vector_type(8)))  __bf16 v8bf;
typedef __attribute__((ext_vector_type(8)))  float  v8f;

// ---------------------------------------------------------------- helpers
__device__ __forceinline__ float bfl(unsigned u) {            // low bf16 -> f32
    return __builtin_bit_cast(float, u << 16);
}
__device__ __forceinline__ float bfh(unsigned u) {            // high bf16 -> f32
    return __builtin_bit_cast(float, u & 0xffff0000u);
}
__device__ __forceinline__ float dot8(uint4 q, const float* h8, float acc) {
    acc = fmaf(bfl(q.x), h8[0], acc);
    acc = fmaf(bfh(q.x), h8[1], acc);
    acc = fmaf(bfl(q.y), h8[2], acc);
    acc = fmaf(bfh(q.y), h8[3], acc);
    acc = fmaf(bfl(q.z), h8[4], acc);
    acc = fmaf(bfh(q.z), h8[5], acc);
    acc = fmaf(bfl(q.w), h8[6], acc);
    acc = fmaf(bfh(q.w), h8[7], acc);
    return acc;
}
__device__ __forceinline__ float sigmoidf_fast(float x) {
    return 1.0f / (1.0f + __expf(-x));
}

// CDNA5 async global->LDS copy (no VGPR round trip, tracked by ASYNCcnt).
// %0 = wave-relative LDS byte offset (low 32 bits of generic LDS pointer,
// per the LDS aperture rule); %1 = 64-bit global address.
__device__ __forceinline__ void async_ld16(const float* g, float* l) {
    unsigned lds_off = (unsigned)(size_t)(void*)l;
    asm volatile("global_load_async_to_lds_b128 %0, %1, off"
                 :: "v"(lds_off), "v"(g)
                 : "memory");
}
__device__ __forceinline__ void async_wait0() {
    asm volatile("s_wait_asynccnt 0x0" ::: "memory");
}

// ---------------------------------------------------------------- fp32 -> bf16 (RNE)
__global__ void cvt_f32_bf16(const float* __restrict__ src,
                             unsigned short* __restrict__ dst, int n) {
    int i = blockIdx.x * blockDim.x + threadIdx.x;
    if (i < n) {
        unsigned u = __builtin_bit_cast(unsigned, src[i]);
        unsigned r = (u + 0x7fffu + ((u >> 16) & 1u)) >> 16;
        dst[i] = (unsigned short)r;
    }
}

__global__ void init_barrier(unsigned* __restrict__ bar) {
    if (threadIdx.x < 8) bar[threadIdx.x] = 0u;
}

// ---------------------------------------------------------------- WMMA projection GEMMs
// X = A[T,E](bf16) @ W[H,E]^T(bf16) -> f32 [T,H]; blockIdx.z picks W_r/W_z/W.
// One wave per 16x16 output tile, K looped in steps of 32 via v_wmma_f32_16x16x32_bf16.
__global__ void __launch_bounds__(THREADS)
proj_gemm(const unsigned short* __restrict__ A,
          const unsigned short* __restrict__ Wr,
          const unsigned short* __restrict__ Wz,
          const unsigned short* __restrict__ Wh,
          float* __restrict__ Xr, float* __restrict__ Xz, float* __restrict__ Xh) {
    const int lane = threadIdx.x & 31;
    const int wave = threadIdx.x >> 5;

    const unsigned short* Wm = (blockIdx.z == 0) ? Wr : (blockIdx.z == 1) ? Wz : Wh;
    float* O = (blockIdx.z == 0) ? Xr : (blockIdx.z == 1) ? Xz : Xh;

    const int tileN = (blockIdx.x * 8 + wave) * 16;
    const int tileM = blockIdx.y * 16;

    // A fragment (16x32, MxK): lane<16 -> M=lane, K 0..7 & 16..23 ; lane>=16 -> K 8..15 & 24..31
    const int mrow = tileM + (lane & 15);
    const int kA   = (lane >> 4) * 8;
    // B fragment (32x16, KxN): lane holds column N=lane&15, same K split as A
    const int ncol = tileN + (lane & 15);
    const int kB   = (lane >> 4) * 8;

    const unsigned short* aptr = A  + (size_t)mrow * EE + kA;
    const unsigned short* bptr = Wm + (size_t)ncol * EE + kB;

    v8f acc = {};
#pragma unroll 4
    for (int k0 = 0; k0 < EE; k0 += 32) {
        __builtin_prefetch(aptr + k0 + 128, 0, 1);
        __builtin_prefetch(bptr + k0 + 128, 0, 1);
        v8bf alo = *(const v8bf*)(aptr + k0);
        v8bf ahi = *(const v8bf*)(aptr + k0 + 16);
        v16bf a;
#pragma unroll
        for (int i = 0; i < 8; ++i) { a[i] = alo[i]; a[i + 8] = ahi[i]; }
        v8bf blo = *(const v8bf*)(bptr + k0);
        v8bf bhi = *(const v8bf*)(bptr + k0 + 16);
        v16bf b;
#pragma unroll
        for (int i = 0; i < 8; ++i) { b[i] = blo[i]; b[i + 8] = bhi[i]; }
        acc = __builtin_amdgcn_wmma_f32_16x16x32_bf16(
            /*neg_a=*/false, a, /*neg_b=*/false, b,
            /*c_mod=*/(short)0, acc, /*reuse_a=*/false, /*reuse_b=*/false);
    }

    // D layout: VGPR r holds M = r + 8*(lane>=16), N = lane&15
    const int mbase = tileM + ((lane >> 4) << 3);
    const int nout  = tileN + (lane & 15);
#pragma unroll
    for (int r = 0; r < 8; ++r) {
        O[(size_t)(mbase + r) * HH + nout] = acc[r];
    }
}

// ---------------------------------------------------------------- grid-wide barrier
__device__ __forceinline__ void grid_barrier(unsigned* cnt, unsigned* gen) {
    __threadfence();
    __syncthreads();
    if (threadIdx.x == 0) {
        unsigned g = __hip_atomic_load(gen, __ATOMIC_ACQUIRE, __HIP_MEMORY_SCOPE_AGENT);
        unsigned arrived =
            __hip_atomic_fetch_add(cnt, 1u, __ATOMIC_ACQ_REL, __HIP_MEMORY_SCOPE_AGENT) + 1u;
        if (arrived == (unsigned)NBLK) {
            __hip_atomic_store(cnt, 0u, __ATOMIC_RELEASE, __HIP_MEMORY_SCOPE_AGENT);
            __hip_atomic_fetch_add(gen, 1u, __ATOMIC_ACQ_REL, __HIP_MEMORY_SCOPE_AGENT);
        } else {
            while (__hip_atomic_load(gen, __ATOMIC_ACQUIRE, __HIP_MEMORY_SCOPE_AGENT) == g) {
                __builtin_amdgcn_s_sleep(8);
            }
        }
    }
    __syncthreads();
}

// ---------------------------------------------------------------- persistent recurrent scan
// 64 blocks x 256 threads; block owns 32 rows; 8 threads per row, 256-wide K chunks.
// U matrices stay L2-resident in bf16 (24 MB << 192 MB L2).  h and r*h staged into
// LDS per phase via CDNA5 async global->LDS B128 copies (ASYNCcnt).
__global__ void __launch_bounds__(THREADS)
gru_scan(const float* __restrict__ h0,
         const unsigned short* __restrict__ Ur,
         const unsigned short* __restrict__ Uz,
         const unsigned short* __restrict__ Uh,
         const float* __restrict__ Xr,
         const float* __restrict__ Xz,
         const float* __restrict__ Xh,
         float* __restrict__ rh,
         unsigned* __restrict__ bar,
         float* __restrict__ out) {
    __shared__ float sh[HH];   // previous hidden state
    __shared__ float sv[HH];   // r * h vector

    const int tid   = threadIdx.x;
    const int chunk = tid & 7;        // 8 threads cooperate per row
    const int row   = tid >> 3;       // 0..31
    const int grow  = blockIdx.x * ROWS + row;
    const int kbase = chunk * (HH / 8);   // 256 contiguous K per thread

    const uint4* urp = (const uint4*)(Ur + (size_t)grow * HH + kbase);
    const uint4* uzp = (const uint4*)(Uz + (size_t)grow * HH + kbase);
    const uint4* uhp = (const uint4*)(Uh + (size_t)grow * HH + kbase);

    float zval = 0.0f;   // lives in a register across the mid-step barrier (chunk==0 lanes)

    for (int t = 0; t < TT; ++t) {
        // ---- stage h_{t-1} into LDS via async global->LDS (2 x B128 per thread)
        const float* hprev = (t == 0) ? h0 : (out + (size_t)(t - 1) * HH);
#pragma unroll
        for (int it = 0; it < HH / (THREADS * 4); ++it) {
            int f = (tid + it * THREADS) * 4;
            async_ld16(hprev + f, &sh[f]);
        }
        async_wait0();
        __syncthreads();

        // ---- phase A: r and z gate matvecs
        float ar = 0.0f, az = 0.0f;
        const float* hp = &sh[kbase];
#pragma unroll 4
        for (int k = 0; k < HH / 64; ++k) {      // 32 iters x 8 elems
            uint4 qr = urp[k];
            uint4 qz = uzp[k];
            const float* h8 = hp + k * 8;
            ar = dot8(qr, h8, ar);
            az = dot8(qz, h8, az);
        }
#pragma unroll
        for (int off = 4; off > 0; off >>= 1) {
            ar += __shfl_xor(ar, off, 8);
            az += __shfl_xor(az, off, 8);
        }
        if (chunk == 0) {
            float r = sigmoidf_fast(ar + Xr[(size_t)t * HH + grow]);
            zval    = sigmoidf_fast(az + Xz[(size_t)t * HH + grow]);
            rh[grow] = r * sh[grow];
        }
        grid_barrier(bar, bar + 1);

        // ---- phase B: stage r*h via async global->LDS, then candidate state
#pragma unroll
        for (int it = 0; it < HH / (THREADS * 4); ++it) {
            int f = (tid + it * THREADS) * 4;
            async_ld16(rh + f, &sv[f]);
        }
        async_wait0();
        __syncthreads();

        float ah = 0.0f;
        const float* vp = &sv[kbase];
#pragma unroll 4
        for (int k = 0; k < HH / 64; ++k) {
            uint4 qh = uhp[k];
            ah = dot8(qh, vp + k * 8, ah);
        }
#pragma unroll
        for (int off = 4; off > 0; off >>= 1) ah += __shfl_xor(ah, off, 8);

        if (chunk == 0) {
            float ht = tanhf(ah + Xh[(size_t)t * HH + grow]);
            out[(size_t)t * HH + grow] = (1.0f - zval) * sh[grow] + zval * ht;
        }
        grid_barrier(bar, bar + 1);
    }
}

// ---------------------------------------------------------------- host entry
extern "C" void kernel_launch(void* const* d_in, const int* in_sizes, int n_in,
                              void* d_out, int out_size, void* d_ws, size_t ws_size,
                              hipStream_t stream) {
    (void)in_sizes; (void)n_in; (void)out_size; (void)ws_size;

    const float* inputs = (const float*)d_in[0];
    const float* h0     = (const float*)d_in[1];
    const float* U_r    = (const float*)d_in[2];
    const float* W_r    = (const float*)d_in[3];
    const float* U_z    = (const float*)d_in[4];
    const float* W_z    = (const float*)d_in[5];
    const float* U      = (const float*)d_in[6];
    const float* W      = (const float*)d_in[7];
    float* out = (float*)d_out;

    char* ws = (char*)d_ws;
    size_t off = 0;
    auto alloc = [&](size_t bytes) -> void* {
        off = (off + 255) & ~(size_t)255;
        void* p = ws + off;
        off += bytes;
        return p;
    };

    unsigned short* Abf = (unsigned short*)alloc((size_t)TT * EE * 2);
    unsigned short* Wrb = (unsigned short*)alloc((size_t)HH * EE * 2);
    unsigned short* Wzb = (unsigned short*)alloc((size_t)HH * EE * 2);
    unsigned short* Whb = (unsigned short*)alloc((size_t)HH * EE * 2);
    unsigned short* Urb = (unsigned short*)alloc((size_t)HH * HH * 2);
    unsigned short* Uzb = (unsigned short*)alloc((size_t)HH * HH * 2);
    unsigned short* Uhb = (unsigned short*)alloc((size_t)HH * HH * 2);
    float* Xr = (float*)alloc((size_t)TT * HH * 4);
    float* Xz = (float*)alloc((size_t)TT * HH * 4);
    float* Xh = (float*)alloc((size_t)TT * HH * 4);
    float* rh = (float*)alloc((size_t)HH * 4);
    unsigned* bar = (unsigned*)alloc(256);

    // fp32 -> bf16 conversions
    {
        int nA = TT * EE;
        cvt_f32_bf16<<<(nA + 255) / 256, 256, 0, stream>>>(inputs, Abf, nA);
        int nW = HH * EE;
        cvt_f32_bf16<<<(nW + 255) / 256, 256, 0, stream>>>(W_r, Wrb, nW);
        cvt_f32_bf16<<<(nW + 255) / 256, 256, 0, stream>>>(W_z, Wzb, nW);
        cvt_f32_bf16<<<(nW + 255) / 256, 256, 0, stream>>>(W,   Whb, nW);
        int nU = HH * HH;
        cvt_f32_bf16<<<(nU + 255) / 256, 256, 0, stream>>>(U_r, Urb, nU);
        cvt_f32_bf16<<<(nU + 255) / 256, 256, 0, stream>>>(U_z, Uzb, nU);
        cvt_f32_bf16<<<(nU + 255) / 256, 256, 0, stream>>>(U,   Uhb, nU);
    }

    init_barrier<<<1, 32, 0, stream>>>(bar);

    // three WMMA input-projection GEMMs
    dim3 gg(HH / (16 * 8), TT / 16, 3);   // (16, 256, 3)
    proj_gemm<<<gg, THREADS, 0, stream>>>(Abf, Wrb, Wzb, Whb, Xr, Xz, Xh);

    // persistent recurrent scan
    gru_scan<<<NBLK, THREADS, 0, stream>>>(h0, Urb, Uzb, Uhb, Xr, Xz, Xh, rh, bar, out);
}
